// SFTGNN_21552145891816
// MI455X (gfx1250) — compile-verified, matrix operand order checked by reading, not run on hardware
//
#include <hip/hip_runtime.h>
#include <hip/hip_bf16.h>
#include <string.h>

// ---------------------------------------------------------------------------
// SFT-GNN on MI455X (gfx1250). Dominant edge-message GEMMs run on
// v_wmma_f32_16x16x32_bf16 (f32 accumulate); z-gather uses the CDNA5
// async-to-LDS engine (global_load_async_to_lds_b128, ASYNCcnt-tracked);
// small MLPs stay f32 VALU.
// ---------------------------------------------------------------------------

typedef __attribute__((ext_vector_type(16))) __bf16 v16bf;
typedef __attribute__((ext_vector_type(8)))  float  v8f;

#define NFEAT 128
#define ZDIM  384
#define ZROW  392            // padded LDS row (shorts): 392*2B=784B -> 4-bank rotation/row
#define KTILES 12            // 384 / 32
#define NTILES 8             // 128 / 16
#define EPB   32             // edges per block (two 16-row M tiles)

union FragU {
    uint4 q[2];
    v16bf v;
};

__device__ __forceinline__ unsigned short f2bf(float f) {
    unsigned int u = __float_as_uint(f);
    unsigned int r = u + 0x7fffu + ((u >> 16) & 1u);      // RNE
    if ((u & 0x7f800000u) == 0x7f800000u) r = u;          // keep inf/nan payload
    return (unsigned short)(r >> 16);
}

__device__ __forceinline__ float siluf(float x) {
    return x / (1.0f + __expf(-x));
}

// fast, numerically stable softplus: max(x,0) + log1p(exp(-|x|)) via HW exp/log
__device__ __forceinline__ float softplusf_fast(float x) {
    return fmaxf(x, 0.0f) + __logf(1.0f + __expf(-fabsf(x)));
}

// ---------------------------------------------------------------------------
// Pack Wg/Wc (f32, [5][384][128]) into per-lane-contiguous bf16 WMMA B-frags:
// dst[mat][l][nt][kt][lane][16], element j of lane L covers
//   k = kt*32 + (L<16?0:8) + (j&7) + (j>=8?16:0),  n = nt*16 + (L&15)
// ---------------------------------------------------------------------------
__global__ void pack_weights_k(const float* __restrict__ Wg,
                               const float* __restrict__ Wc,
                               unsigned short* __restrict__ fragG,
                               unsigned short* __restrict__ fragC) {
    int t = blockIdx.x * blockDim.x + threadIdx.x;
    const int total = 2 * 5 * NTILES * KTILES * 32 * 16;
    if (t >= total) return;
    int j    = t & 15;  t >>= 4;
    int lane = t & 31;  t >>= 5;
    int kt   = t % KTILES; t /= KTILES;
    int nt   = t & 7;   t >>= 3;
    int l    = t % 5;
    int mat  = t / 5;

    int k = kt * 32 + ((lane < 16) ? 0 : 8) + (j & 7) + ((j >= 8) ? 16 : 0);
    int n = nt * 16 + (lane & 15);
    const float* W = (mat == 0) ? Wg : Wc;
    float v = W[((size_t)l * ZDIM + k) * NFEAT + n];
    size_t dst = ((((size_t)l * NTILES + nt) * KTILES + kt) * 32 + lane) * 16 + j;
    unsigned short* out = (mat == 0) ? fragG : fragC;
    out[dst] = f2bf(v);
}

// ---------------------------------------------------------------------------
// Atom embedding: h = x @ W_emb + b  ([N,92] -> [N,128]), + bf16 mirror.
// ---------------------------------------------------------------------------
__global__ void embed_k(const float* __restrict__ x,
                        const float* __restrict__ W,
                        const float* __restrict__ b,
                        float* __restrict__ h,
                        unsigned short* __restrict__ hbf,
                        int nN) {
    int node = blockIdx.x;
    int f = threadIdx.x;                 // 128 threads
    if (node >= nN) return;
    __shared__ float xr[92];
    if (f < 92) xr[f] = x[(size_t)node * 92 + f];
    __syncthreads();
    float acc = b[f];
    #pragma unroll 4
    for (int i = 0; i < 92; ++i) acc = fmaf(xr[i], W[i * NFEAT + f], acc);
    size_t o = (size_t)node * NFEAT + f;
    h[o] = acc;
    hbf[o] = f2bf(acc);
}

// ---------------------------------------------------------------------------
// Edge features: edge_attr = [rbf-MLP(1/d) (64) | sh-MLP(sh) (64)] -> bf16.
// One block of 128 threads per edge.
// ---------------------------------------------------------------------------
__global__ void edge_attr_k(const float* __restrict__ sh,
                            const float* __restrict__ edge_dist,
                            const float* __restrict__ shW1, const float* __restrict__ shB1,
                            const float* __restrict__ shW2, const float* __restrict__ shB2,
                            const float* __restrict__ rbW1, const float* __restrict__ rbB1,
                            const float* __restrict__ rbW2, const float* __restrict__ rbB2,
                            unsigned short* __restrict__ eabf,
                            int nE) {
    int e = blockIdx.x;
    int t = threadIdx.x;                 // 128 threads
    if (e >= nE) return;
    __shared__ float sh9[9];
    __shared__ float rbfv[64];
    __shared__ float hid32[32];
    __shared__ float rh[64];

    if (t < 9) sh9[t] = sh[(size_t)e * 9 + t];
    if (t < 64) {
        const float vmin = 0.125f, vmax = 1.4f;
        const float step = (vmax - vmin) / 63.0f;
        const float gamma = 1.0f / (step * step);
        float invd = 1.0f / edge_dist[e];
        float c = vmin + step * (float)t;
        float d = invd - c;
        rbfv[t] = __expf(-gamma * d * d);
    }
    __syncthreads();
    if (t < 32) {
        float a = shB1[t];
        #pragma unroll
        for (int i = 0; i < 9; ++i) a = fmaf(sh9[i], shW1[i * 32 + t], a);
        hid32[t] = siluf(a);
    }
    if (t >= 64) {   // rbf hidden layer on threads 64..127 mapped to units 0..63
        int u = t - 64;
        float a = rbB1[u];
        #pragma unroll 4
        for (int k = 0; k < 64; ++k) a = fmaf(rbfv[k], rbW1[k * 64 + u], a);
        rh[u] = siluf(a);
    }
    __syncthreads();
    if (t < 64) {            // dist_e -> cols [0,64)
        float a = rbB2[t];
        #pragma unroll 4
        for (int k = 0; k < 64; ++k) a = fmaf(rh[k], rbW2[k * 64 + t], a);
        eabf[(size_t)e * NFEAT + t] = f2bf(a);
    } else {                 // sh_e -> cols [64,128)
        int j = t - 64;
        float a = shB2[j];
        #pragma unroll
        for (int i = 0; i < 32; ++i) a = fmaf(hid32[i], shW2[i * 64 + j], a);
        eabf[(size_t)e * NFEAT + 64 + j] = f2bf(a);
    }
}

// ---------------------------------------------------------------------------
// Edge message GEMM + gated activation + scatter-add. 32 edges/block,
// 8 waves; wave w owns output columns 16w..16w+15 and accumulates 4 WMMA
// chains (2 M-tiles x {Wg,Wc}) over 12 K-steps of 32 (bf16 -> f32).
// z-gather goes straight to LDS through the async-to-LDS engine (ASYNCcnt).
// ---------------------------------------------------------------------------
__global__ void __launch_bounds__(256, 2)
edge_msg_k(const unsigned short* __restrict__ hbf,
           const unsigned short* __restrict__ eabf,
           const int* __restrict__ srcp,
           const int* __restrict__ dstp,
           const unsigned short* __restrict__ wG,   // layer-offset frag ptr
           const unsigned short* __restrict__ wC,
           const float* __restrict__ bgl,
           const float* __restrict__ bcl,
           float* __restrict__ agg,
           int nE) {
    __shared__ __align__(16) unsigned short zt[EPB * ZROW];
    __shared__ int dstid[EPB];

    const int tid = threadIdx.x;
    const int eb = blockIdx.x * EPB;

    // ---- Phase 1: gather z = [h[dst] | h[src] | edge_attr] as bf16 tiles ----
    // Each 16B chunk: global -> LDS directly, no VGPR staging.
    for (int c = tid; c < EPB * 48; c += 256) {
        int e = c / 48;
        int off = c - e * 48;            // 16B chunk index within the 384-feat row
        int ge = eb + e;
        unsigned short* lptr = &zt[e * ZROW + off * 8];
        if (ge < nE) {
            const unsigned short* g;
            if (off < 16)
                g = hbf + (size_t)dstp[ge] * NFEAT + off * 8;
            else if (off < 32)
                g = hbf + (size_t)srcp[ge] * NFEAT + (off - 16) * 8;
            else
                g = eabf + (size_t)ge * NFEAT + (off - 32) * 8;
            // LDS byte offset = low 32 bits of the generic shared-aperture addr
            unsigned lds = (unsigned)(size_t)lptr;
            asm volatile("global_load_async_to_lds_b128 %0, %1, off"
                         :: "v"(lds), "v"(g) : "memory");
        } else {
            *(uint4*)lptr = make_uint4(0u, 0u, 0u, 0u);
        }
    }
    if (tid < EPB) dstid[tid] = (eb + tid < nE) ? dstp[eb + tid] : 0;
    asm volatile("s_wait_asynccnt 0" ::: "memory");
    __syncthreads();

    // ---- Phase 2: WMMA ----
    const int wave = tid >> 5;           // 0..7 -> output N-tile
    const int lane = tid & 31;
    const int n0 = wave * 16;
    const int m = lane & 15;
    const int kb = (lane < 16) ? 0 : 8;  // ISA 16-bit A/B fragment K base

    v8f accG0 = {}, accG1 = {}, accC0 = {}, accC1 = {};

    #pragma unroll 2
    for (int kt = 0; kt < KTILES; ++kt) {
        FragU a0, a1, bG, bC;
        const unsigned short* zr0 = &zt[m * ZROW + kt * 32 + kb];
        const unsigned short* zr1 = &zt[(16 + m) * ZROW + kt * 32 + kb];
        a0.q[0] = *(const uint4*)zr0;        a0.q[1] = *(const uint4*)(zr0 + 16);
        a1.q[0] = *(const uint4*)zr1;        a1.q[1] = *(const uint4*)(zr1 + 16);
        const unsigned short* wg = wG + ((size_t)(wave * KTILES + kt) * 32 + lane) * 16;
        const unsigned short* wc = wC + ((size_t)(wave * KTILES + kt) * 32 + lane) * 16;
        bG.q[0] = *(const uint4*)wg;         bG.q[1] = *(const uint4*)(wg + 8);
        bC.q[0] = *(const uint4*)wc;         bC.q[1] = *(const uint4*)(wc + 8);

        accG0 = __builtin_amdgcn_wmma_f32_16x16x32_bf16(false, a0.v, false, bG.v,
                                                        (short)0, accG0, false, false);
        accG1 = __builtin_amdgcn_wmma_f32_16x16x32_bf16(false, a1.v, false, bG.v,
                                                        (short)0, accG1, false, false);
        accC0 = __builtin_amdgcn_wmma_f32_16x16x32_bf16(false, a0.v, false, bC.v,
                                                        (short)0, accC0, false, false);
        accC1 = __builtin_amdgcn_wmma_f32_16x16x32_bf16(false, a1.v, false, bC.v,
                                                        (short)0, accC1, false, false);
    }

    // ---- Epilogue: bias + sigmoid*softplus, scatter-add into agg[dst] ----
    const int n = n0 + (lane & 15);
    const int rowbase = (lane < 16) ? 0 : 8;     // C/D layout: vgpr r -> M=r(+8)
    const float bgv = bgl[n];
    const float bcv = bcl[n];

    #pragma unroll
    for (int mt = 0; mt < 2; ++mt) {
        #pragma unroll
        for (int r = 0; r < 8; ++r) {
            float g = ((mt == 0) ? accG0[r] : accG1[r]) + bgv;
            float c = ((mt == 0) ? accC0[r] : accC1[r]) + bcv;
            float sg = 1.0f / (1.0f + __expf(-g));
            float sp = softplusf_fast(c);
            int er = mt * 16 + rowbase + r;
            if (eb + er < nE)
                atomicAdd(&agg[(size_t)dstid[er] * NFEAT + n], sg * sp);
        }
    }
}

// ---------------------------------------------------------------------------
// h += agg; refresh bf16 mirror.
// ---------------------------------------------------------------------------
__global__ void update_h_k(float* __restrict__ h,
                           const float* __restrict__ agg,
                           unsigned short* __restrict__ hbf,
                           size_t n) {
    size_t i = (size_t)blockIdx.x * blockDim.x + threadIdx.x;
    if (i < n) {
        float v = h[i] + agg[i];
        h[i] = v;
        hbf[i] = f2bf(v);
    }
}

// ---------------------------------------------------------------------------
// Scatter-mean pooling (phase 1: sums + counts).
// ---------------------------------------------------------------------------
__global__ void pool_k(const float* __restrict__ h,
                       const int* __restrict__ batch,
                       float* __restrict__ sums,
                       float* __restrict__ counts,
                       int nN) {
    int node = blockIdx.x;
    int f = threadIdx.x;                 // 128 threads
    if (node >= nN) return;
    int b = batch[node];
    atomicAdd(&sums[(size_t)b * NFEAT + f], h[(size_t)node * NFEAT + f]);
    if (f == 0) atomicAdd(&counts[b], 1.0f);
}

// ---------------------------------------------------------------------------
// Readout MLP: pooled -> silu(fc1) -> fc2 -> out[g]. One block per graph.
// ---------------------------------------------------------------------------
__global__ void readout_k(const float* __restrict__ sums,
                          const float* __restrict__ counts,
                          const float* __restrict__ W1, const float* __restrict__ b1,
                          const float* __restrict__ W2, const float* __restrict__ b2,
                          float* __restrict__ out,
                          int nG) {
    int g = blockIdx.x;
    int t = threadIdx.x;                 // 128 threads
    if (g >= nG) return;
    __shared__ float pooled[NFEAT];
    __shared__ float red[NFEAT];
    float cnt = fmaxf(counts[g], 1.0f);
    pooled[t] = sums[(size_t)g * NFEAT + t] / cnt;
    __syncthreads();
    float a = b1[t];
    #pragma unroll 4
    for (int i = 0; i < NFEAT; ++i) a = fmaf(pooled[i], W1[i * NFEAT + t], a);
    red[t] = siluf(a) * W2[t];
    __syncthreads();
    for (int s = 64; s > 0; s >>= 1) {
        if (t < s) red[t] += red[t + s];
        __syncthreads();
    }
    if (t == 0) out[g] = red[0] + b2[0];
}

// ---------------------------------------------------------------------------
// Host-side orchestration.
// ---------------------------------------------------------------------------
static inline size_t align256(size_t x) { return (x + 255) & ~(size_t)255; }

extern "C" void kernel_launch(void* const* d_in, const int* in_sizes, int n_in,
                              void* d_out, int out_size, void* d_ws, size_t ws_size,
                              hipStream_t stream) {
    (void)n_in; (void)ws_size;
    // ---- inputs (setup_inputs dict order) ----
    const float* x         = (const float*)d_in[0];
    const int*   ei        = (const int*)d_in[1];
    const float* sh        = (const float*)d_in[2];
    const float* edist     = (const float*)d_in[3];
    const int*   batch     = (const int*)d_in[4];
    const float* W_emb     = (const float*)d_in[5];
    const float* b_emb     = (const float*)d_in[6];
    const float* sh_W1     = (const float*)d_in[7];
    const float* sh_b1     = (const float*)d_in[8];
    const float* sh_W2     = (const float*)d_in[9];
    const float* sh_b2     = (const float*)d_in[10];
    const float* rbf_W1    = (const float*)d_in[11];
    const float* rbf_b1    = (const float*)d_in[12];
    const float* rbf_W2    = (const float*)d_in[13];
    const float* rbf_b2    = (const float*)d_in[14];
    const float* Wg        = (const float*)d_in[15];
    const float* bg        = (const float*)d_in[16];
    const float* Wc        = (const float*)d_in[17];
    const float* bc        = (const float*)d_in[18];
    const float* fc_W1     = (const float*)d_in[19];
    const float* fc_b1     = (const float*)d_in[20];
    const float* fc_W2     = (const float*)d_in[21];
    const float* fc_b2     = (const float*)d_in[22];
    float* out = (float*)d_out;

    const int nN = in_sizes[0] / 92;         // 50000
    const int nE = in_sizes[3];              // 500000
    const int nG = out_size;                 // 1024
    const int nL = in_sizes[16] / NFEAT;     // 5 layers
    const int* srcp = ei;
    const int* dstp = ei + nE;

    // ---- workspace carve-up ----
    char* ws = (char*)d_ws;
    size_t off = 0;
    float* h      = (float*)(ws + off);        off = align256(off + (size_t)nN * NFEAT * 4);
    float* agg    = (float*)(ws + off);        off = align256(off + (size_t)nN * NFEAT * 4);
    unsigned short* hbf  = (unsigned short*)(ws + off); off = align256(off + (size_t)nN * NFEAT * 2);
    unsigned short* eabf = (unsigned short*)(ws + off); off = align256(off + (size_t)nE * NFEAT * 2);
    const size_t fragPerLayer = (size_t)NTILES * KTILES * 32 * 16;   // shorts
    unsigned short* fragG = (unsigned short*)(ws + off); off = align256(off + nL * fragPerLayer * 2);
    unsigned short* fragC = (unsigned short*)(ws + off); off = align256(off + nL * fragPerLayer * 2);
    float* sums   = (float*)(ws + off);        off = align256(off + (size_t)nG * NFEAT * 4);
    float* counts = (float*)(ws + off);        off = align256(off + (size_t)nG * 4);

    // ---- 0. swizzle weights into WMMA B-fragment order ----
    {
        int total = 2 * nL * NTILES * KTILES * 32 * 16;
        pack_weights_k<<<(total + 255) / 256, 256, 0, stream>>>(Wg, Wc, fragG, fragC);
    }
    // ---- 1. atom embedding ----
    embed_k<<<nN, NFEAT, 0, stream>>>(x, W_emb, b_emb, h, hbf, nN);
    // ---- 2. edge features ----
    edge_attr_k<<<nE, NFEAT, 0, stream>>>(sh, edist, sh_W1, sh_b1, sh_W2, sh_b2,
                                          rbf_W1, rbf_b1, rbf_W2, rbf_b2, eabf, nE);
    // ---- 3. message-passing layers ----
    const int msgBlocks = (nE + EPB - 1) / EPB;
    const size_t hElems = (size_t)nN * NFEAT;
    for (int l = 0; l < nL; ++l) {
        hipMemsetAsync(agg, 0, hElems * sizeof(float), stream);
        edge_msg_k<<<msgBlocks, 256, 0, stream>>>(
            hbf, eabf, srcp, dstp,
            fragG + (size_t)l * fragPerLayer,
            fragC + (size_t)l * fragPerLayer,
            bg + (size_t)l * NFEAT, bc + (size_t)l * NFEAT,
            agg, nE);
        update_h_k<<<(int)((hElems + 255) / 256), 256, 0, stream>>>(h, agg, hbf, hElems);
    }
    // ---- 4. scatter-mean readout ----
    hipMemsetAsync(sums, 0, (size_t)nG * NFEAT * sizeof(float), stream);
    hipMemsetAsync(counts, 0, (size_t)nG * sizeof(float), stream);
    pool_k<<<nN, NFEAT, 0, stream>>>(h, batch, sums, counts, nN);
    readout_k<<<nG, NFEAT, 0, stream>>>(sums, counts, fc_W1, fc_b1, fc_W2, fc_b2, out, nG);
}
